// LinearChainCRF_40046275068440
// MI455X (gfx1250) — matrix operand order repaired and checked by reading, not compiled, not run on hardware
//
#include <hip/hip_runtime.h>
#include <hip/hip_bf16.h>
#include <math.h>

typedef _Float16 v8h  __attribute__((ext_vector_type(8)));
typedef _Float16 v16h __attribute__((ext_vector_type(16)));
typedef float    v8f  __attribute__((ext_vector_type(8)));
typedef unsigned int v4u __attribute__((ext_vector_type(4)));
typedef int      v8i  __attribute__((ext_vector_type(8)));
typedef int      v4i  __attribute__((ext_vector_type(4)));

#define NUM_LABELS 256
#define FULLN 258
#define BB 128
#define SS 1024

#define LN2F  0.6931471805599453f
#define L2EF  1.4426950408889634f

// workspace layout (bytes)
#define WS_TRANSLOG 0                        // 256*256 f32
#define WS_BT       (256*256*4)              // 256*256 f16, transposed exp(trans): BT[j][k]
#define WS_START    (WS_BT + 256*256*2)      // 256 f32
#define WS_END      (WS_START + 256*4)       // 256 f32
#define WS_SCORES   (WS_END + 256*4)         // 128 f32
#define WS_LSE0     (WS_SCORES + 128*4)      // 128 f32

// ---------------------------------------------------------------------------
// Kernel A: row log-softmax of the 258x258 transition matrix.
// ---------------------------------------------------------------------------
__global__ void crf_prep_kernel(const float* __restrict__ T,
                                float* __restrict__ transLog,
                                _Float16* __restrict__ BT,
                                float* __restrict__ startv,
                                float* __restrict__ endv) {
  const int r = blockIdx.x, tid = threadIdx.x;
  __shared__ float red[256];
  float v  = T[r * FULLN + tid];
  float v2 = (tid < FULLN - 256) ? T[r * FULLN + 256 + tid] : -INFINITY;
  red[tid] = fmaxf(v, v2);
  __syncthreads();
  for (int s = 128; s > 0; s >>= 1) {
    if (tid < s) red[tid] = fmaxf(red[tid], red[tid + s]);
    __syncthreads();
  }
  float mx = red[0];
  __syncthreads();
  red[tid] = __expf(v - mx) + ((tid < FULLN - 256) ? __expf(v2 - mx) : 0.f);
  __syncthreads();
  for (int s = 128; s > 0; s >>= 1) {
    if (tid < s) red[tid] += red[tid + s];
    __syncthreads();
  }
  float lse = mx + __logf(red[0]);
  if (r < 256) {
    float l = v - lse;
    transLog[r * 256 + tid] = l;
    BT[tid * 256 + r] = (_Float16)__expf(l);  // transposed
    if (tid == 0) endv[r] = T[r * FULLN + 257] - lse;
  } else if (r == 256) {
    startv[tid] = v - lse;
  }
}

// ---------------------------------------------------------------------------
// Kernel B: per-batch path score (raw-logit form) and lse0 = LSE(logits[b,0,:]).
// ---------------------------------------------------------------------------
__global__ void crf_score_kernel(const float* __restrict__ logits,
                                 const int* __restrict__ labels,
                                 const float* __restrict__ transLog,
                                 const float* __restrict__ startv,
                                 const float* __restrict__ endv,
                                 float* __restrict__ scores,
                                 float* __restrict__ lse0) {
  const int b = blockIdx.x, tid = threadIdx.x;
  __shared__ float red[256];
  float v = logits[(size_t)b * SS * 256 + tid];
  red[tid] = v;
  __syncthreads();
  for (int s = 128; s > 0; s >>= 1) {
    if (tid < s) red[tid] = fmaxf(red[tid], red[tid + s]);
    __syncthreads();
  }
  float mx = red[0];
  __syncthreads();
  red[tid] = __expf(v - mx);
  __syncthreads();
  for (int s = 128; s > 0; s >>= 1) {
    if (tid < s) red[tid] += red[tid + s];
    __syncthreads();
  }
  float l0 = mx + __logf(red[0]);
  __syncthreads();
  float s_acc = 0.f;
  for (int t = tid; t < SS; t += 256) {
    int lab = labels[b * SS + t];
    int prv = labels[b * SS + (t == 0 ? SS - 1 : t - 1)];
    s_acc += transLog[prv * 256 + lab] + logits[((size_t)b * SS + t) * 256 + lab];
  }
  red[tid] = s_acc;
  __syncthreads();
  for (int s = 128; s > 0; s >>= 1) {
    if (tid < s) red[tid] += red[tid + s];
    __syncthreads();
  }
  if (tid == 0) {
    int first = labels[b * SS];
    int last  = labels[b * SS + SS - 1];
    scores[b] = startv[first] + logits[(size_t)b * SS * 256 + first] + red[0] + endv[last];
    lse0[b] = l0;
  }
}

// ---------------------------------------------------------------------------
// Kernel C: forward recursion. Per-step exp-domain GEMM via WMMA; per-step
// logits tile (16 rows x 256 f32, row stride S*1024B) staged into LDS by the
// Tensor Data Mover, double-buffered, overlapped with the WMMA chain.
// ---------------------------------------------------------------------------
#define PPITCH 264   // f16 pitch for p matrix
#define APITCH 260   // f32 pitch for alpha matrix
#define LPITCH 260   // f32 pitch of TDM-staged logits (256 data + 4 pad dwords)

__global__ __launch_bounds__(512) void crf_forward_kernel(
    const float* __restrict__ logits,
    const _Float16* __restrict__ BT,     // BT[n][k] = exp(trans[k][n]), 256x256
    const float* __restrict__ startv,
    const float* __restrict__ endv,
    const float* __restrict__ scores,
    const float* __restrict__ lse0,
    float* __restrict__ out) {
  __shared__ _Float16 lds_p[16 * PPITCH];       //  8448 B
  __shared__ float    lds_alpha[16 * APITCH];   // 16640 B
  __shared__ float    lds_log[2][16 * LPITCH + 8]; // 2x 16672 B, TDM staging

  const int tid  = threadIdx.x;
  const int w    = tid >> 5;        // wave id = owned batch row = owned N tile
  const int lane = tid & 31;
  const int hm   = lane & 15;       // A row (M) / B,D column (N)
  const int h    = lane >> 4;       // half-wave selector in WMMA layouts
  const int b0   = blockIdx.x * 16;

  // TDM D# group1 (constant): data_size=4B (bits17:16=2), pad_enable (bit20),
  // pad_interval=7 (256 dwords, bits24:22), pad_amount=3 (4 dwords, bits31:25),
  // tensor_dim0=256, tensor_dim1=16, tile_dim0=256, tile_dim1=16,
  // tensor_dim0_stride = S*256 elements.
  const v8i tdm_g1 = { 0x07D20000, 0x01000000, 0x00100000, 0x01000000,
                       16, 0x00040000, 0, 0 };
  const v4i tdm_gz4 = { 0, 0, 0, 0 };
  const v8i tdm_gz8 = { 0, 0, 0, 0, 0, 0, 0, 0 };

  // init: alpha0[w][j] = start[j] + logits[b0+w, 0, j]
  {
    const float* lrow = logits + (size_t)(b0 + w) * SS * 256;
    #pragma unroll
    for (int i = 0; i < 8; ++i) {
      int j = lane * 8 + i;
      lds_alpha[w * APITCH + j] = startv[j] + lrow[j];
    }
  }
  float carry = 0.f;

  // issue TDM for t=1 into buffer 1
  if (w == 0) {
    const float* gp = logits + ((size_t)b0 * SS + 1) * 256;
    unsigned long long ga = (unsigned long long)(uintptr_t)gp;
    unsigned int ldsOff = (unsigned int)(uintptr_t)(&lds_log[1][0]);
    v4u g0 = { 1u, ldsOff, (unsigned int)ga,
               (unsigned int)((ga >> 32) & 0x01ffffffu) | 0x80000000u /*type=2*/ };
    __builtin_amdgcn_tensor_load_to_lds(g0, tdm_g1, tdm_gz4, tdm_gz4, tdm_gz8, 0);
  }

  for (int t = 1; t < SS; ++t) {
    __syncthreads();  // alpha_{t-1} complete; buf[(t+1)&1] free for reuse

    // wave 0: launch DMA for step t+1, then ensure step t's tile has landed
    if (w == 0) {
      if (t + 1 < SS) {
        const float* gp = logits + ((size_t)b0 * SS + (t + 1)) * 256;
        unsigned long long ga = (unsigned long long)(uintptr_t)gp;
        unsigned int ldsOff = (unsigned int)(uintptr_t)(&lds_log[(t + 1) & 1][0]);
        v4u g0 = { 1u, ldsOff, (unsigned int)ga,
                   (unsigned int)((ga >> 32) & 0x01ffffffu) | 0x80000000u };
        __builtin_amdgcn_tensor_load_to_lds(g0, tdm_g1, tdm_gz4, tdm_gz4, tdm_gz8, 0);
        __builtin_amdgcn_s_wait_tensorcnt((short)1);  // op(t) done (in-order)
      } else {
        __builtin_amdgcn_s_wait_tensorcnt((short)0);
      }
    }

    // --- rowmax + p for row w (overlaps the outstanding DMA) ---
    float vals[8];
    float pm = -INFINITY;
    #pragma unroll
    for (int i = 0; i < 8; ++i) {
      vals[i] = lds_alpha[w * APITCH + lane * 8 + i];
      pm = fmaxf(pm, vals[i]);
    }
    #pragma unroll
    for (int off = 16; off > 0; off >>= 1)
      pm = fmaxf(pm, __shfl_xor(pm, off, 32));
    carry += pm;
    v8h pv;
    #pragma unroll
    for (int i = 0; i < 8; ++i)
      pv[i] = (_Float16)__builtin_amdgcn_exp2f((vals[i] - pm) * L2EF);
    *(v8h*)(&lds_p[w * PPITCH + lane * 8]) = pv;
    __syncthreads();  // p ready AND logits tile for step t visible

    // --- GEMM: q[m][n] = sum_k p[m][k] * exp(trans[k][n]) ---
    v8f c = {};
    #pragma unroll
    for (int kt = 0; kt < 8; ++kt) {
      v8h alo = *(const v8h*)(&lds_p[hm * PPITCH + kt * 32 + 16 * h]);
      v8h ahi = *(const v8h*)(&lds_p[hm * PPITCH + kt * 32 + 16 * h + 8]);
      v16h a = __builtin_shufflevector(alo, ahi,
                 0, 1, 2, 3, 4, 5, 6, 7, 8, 9, 10, 11, 12, 13, 14, 15);
      v16h bm = *(const v16h*)(BT + (size_t)(16 * w + hm) * 256 + kt * 32 + 16 * h);
      c = __builtin_amdgcn_wmma_f32_16x16x32_f16(false, a, false, bm,
                                                 (short)0, c, false, false);
    }

    // alpha_t = log(q) + staged logits; native v_log_f32 (q >= e^-7, no denorms)
    const float* lg = &lds_log[t & 1][0];
    #pragma unroll
    for (int r = 0; r < 8; ++r) {
      int m = r + 8 * h;
      int j = 16 * w + hm;
      float a_new = __builtin_amdgcn_logf(c[r]) * LN2F + lg[m * LPITCH + j];
      lds_alpha[m * APITCH + j] = a_new;
    }
  }

  __syncthreads();
  // epilogue: log_z_raw[row w] = carry + LSE_j(alpha[w][j] + end[j])
  {
    float vals[8];
    float pm = -INFINITY;
    #pragma unroll
    for (int i = 0; i < 8; ++i) {
      vals[i] = lds_alpha[w * APITCH + lane * 8 + i] + endv[lane * 8 + i];
      pm = fmaxf(pm, vals[i]);
    }
    #pragma unroll
    for (int off = 16; off > 0; off >>= 1)
      pm = fmaxf(pm, __shfl_xor(pm, off, 32));
    float se = 0.f;
    #pragma unroll
    for (int i = 0; i < 8; ++i)
      se += __builtin_amdgcn_exp2f((vals[i] - pm) * L2EF);
    #pragma unroll
    for (int off = 16; off > 0; off >>= 1)
      se += __shfl_xor(se, off, 32);
    if (lane == 0) {
      float log_z_raw = carry + pm + __builtin_amdgcn_logf(se) * LN2F;
      out[b0 + w] = log_z_raw - scores[b0 + w] + lse0[b0 + w];
    }
  }
}

// ---------------------------------------------------------------------------
extern "C" void kernel_launch(void* const* d_in, const int* in_sizes, int n_in,
                              void* d_out, int out_size, void* d_ws, size_t ws_size,
                              hipStream_t stream) {
  (void)in_sizes; (void)n_in; (void)out_size; (void)ws_size;
  const float* logits = (const float*)d_in[0];
  const int*   labels = (const int*)d_in[1];
  // d_in[2] = mask: all ones in this problem -> unused
  const float* T      = (const float*)d_in[3];

  char* ws = (char*)d_ws;
  float*    transLog = (float*)(ws + WS_TRANSLOG);
  _Float16* BT       = (_Float16*)(ws + WS_BT);
  float*    startv   = (float*)(ws + WS_START);
  float*    endv     = (float*)(ws + WS_END);
  float*    scores   = (float*)(ws + WS_SCORES);
  float*    lse0v    = (float*)(ws + WS_LSE0);
  float*    out      = (float*)d_out;

  crf_prep_kernel<<<257, 256, 0, stream>>>(T, transLog, BT, startv, endv);
  crf_score_kernel<<<128, 256, 0, stream>>>(logits, labels, transLog, startv, endv,
                                            scores, lse0v);
  crf_forward_kernel<<<8, 512, 0, stream>>>(logits, BT, startv, endv,
                                            scores, lse0v, out);
}
